// TextToMotionSSM_2662879724371
// MI455X (gfx1250) — matrix operand order repaired and checked by reading, not compiled, not run on hardware
//
#include <hip/hip_runtime.h>

// ---------------------------------------------------------------------------
// TextToMotionSSM forward for gfx1250 (MI455X).
// GEMMs: v_wmma_f32_16x16x32_f16, fragments hand-built from LDS per CDNA5 ISA
// 7.12.2.  Block tile 128x64, 8 waves (4x2), 32x32 per wave = 4 WMMA/K-step.
// Software-pipelined: K-tile k+1 staged to registers while WMMAs consume LDS
// buffer k&1; one barrier per K-step.  Loads are branch-free (row clamping;
// K % 32 == 0 at every call site).  fp32 everywhere else.
// Workspace requirement: ~102 MB (25,440,256 floats).
// ---------------------------------------------------------------------------

typedef __attribute__((ext_vector_type(16))) _Float16 v16h;
typedef __attribute__((ext_vector_type(2)))  __fp16   v2fp;   // cvt_pkrtz return type
typedef __attribute__((ext_vector_type(8)))  float    v8f;

#define OP_NONE     0
#define OP_RELU     1
#define OP_SILU     2
#define OP_SOFTPLUS 3

__device__ __forceinline__ float sigmoidf_(float x) { return 1.f / (1.f + __expf(-x)); }
__device__ __forceinline__ float siluf_(float x)    { return x * sigmoidf_(x); }
__device__ __forceinline__ float softplusf_(float x){ return (x > 20.f) ? x : log1pf(__expf(x)); }

// ---------------------------------------------------------------------------
// C[M,N] = act(A[M,K] @ W[N,K]^T + bias) (+ res).  K % 32 == 0 REQUIRED.
// lda/ldw must be multiples of 4 (for float4 staging loads).
// ---------------------------------------------------------------------------
__global__ __launch_bounds__(256)
void k_gemm(const float* __restrict__ A, const float* __restrict__ W,
            const float* __restrict__ bias, const float* __restrict__ res,
            float* __restrict__ C, int M, int N, int K,
            int lda, int ldw, int ldc, int op)
{
    // row stride 40 halfs = 80B: keeps 8-half segments 16B aligned; 20-dword
    // stride spreads the 16 fragment rows across distinct LDS banks.
    __shared__ _Float16 As[2][128][40];
    __shared__ _Float16 Bs[2][64][40];

    const int tid  = threadIdx.x;
    const int m0   = blockIdx.y * 128;
    const int n0   = blockIdx.x * 64;
    const int lane = tid & 31;
    const int wv   = tid >> 5;      // 0..7
    const int wm   = wv >> 1;       // 0..3  (M sub-block of 32)
    const int wn   = wv & 1;        // 0..1  (N sub-block of 32)
    const int hl   = lane >> 4;     // lane half
    const int lr   = lane & 15;

    // staging coordinates: each thread owns 8 consecutive k of one row
    const int kseg = (tid & 3) * 8;
    const int arow = tid >> 2;      // 0..63 (+64 on second A pass)

    v8f acc00 = {0.f,0.f,0.f,0.f,0.f,0.f,0.f,0.f};
    v8f acc01 = acc00, acc10 = acc00, acc11 = acc00;

    // issue all 6 staging loads before any use -> single clause, one wait
    auto load_tile = [&](int k0, float4 (&r)[6]) {
#pragma unroll
        for (int p = 0; p < 2; ++p) {
            int gm = min(m0 + arow + p * 64, M - 1);   // clamp: OOB rows unstored
            const float4* ap = (const float4*)(A + (long)gm * lda + k0 + kseg);
            r[2 * p + 0] = ap[0];
            r[2 * p + 1] = ap[1];
        }
        int gn = min(n0 + arow, N - 1);                // clamp: OOB cols unstored
        const float4* bp = (const float4*)(W + (long)gn * ldw + k0 + kseg);
        r[4] = bp[0];
        r[5] = bp[1];
    };
    auto store_tile = [&](int buf, const float4 (&r)[6]) {
#pragma unroll
        for (int p = 0; p < 2; ++p) {
            union { v2fp h2[4]; uint4 u4; } pk;
            pk.h2[0] = __builtin_amdgcn_cvt_pkrtz(r[2*p].x, r[2*p].y);
            pk.h2[1] = __builtin_amdgcn_cvt_pkrtz(r[2*p].z, r[2*p].w);
            pk.h2[2] = __builtin_amdgcn_cvt_pkrtz(r[2*p+1].x, r[2*p+1].y);
            pk.h2[3] = __builtin_amdgcn_cvt_pkrtz(r[2*p+1].z, r[2*p+1].w);
            *(uint4*)&As[buf][arow + p * 64][kseg] = pk.u4;
        }
        union { v2fp h2[4]; uint4 u4; } pk;
        pk.h2[0] = __builtin_amdgcn_cvt_pkrtz(r[4].x, r[4].y);
        pk.h2[1] = __builtin_amdgcn_cvt_pkrtz(r[4].z, r[4].w);
        pk.h2[2] = __builtin_amdgcn_cvt_pkrtz(r[5].x, r[5].y);
        pk.h2[3] = __builtin_amdgcn_cvt_pkrtz(r[5].z, r[5].w);
        *(uint4*)&Bs[buf][arow][kseg] = pk.u4;
    };

    const int nk = K >> 5;
    {
        float4 r0[6];
        load_tile(0, r0);
        store_tile(0, r0);
    }
    __syncthreads();

    const int am0 = wm * 32 + lr;
    const int am1 = am0 + 16;
    const int bn0 = wn * 32 + lr;
    const int bn1 = bn0 + 16;

    for (int kk = 0; kk < nk; ++kk) {
        float4 rnxt[6];
        if (kk + 1 < nk) load_tile((kk + 1) << 5, rnxt);          // overlaps WMMAs
        if (kk + 2 < nk) {                                        // L2 prefetch (tile k+2)
            int pm = min(m0 + arow, M - 1);
            __builtin_prefetch(A + (long)pm * lda + ((kk + 2) << 5) + kseg, 0, 1);
        }

        const int buf = kk & 1;
        union { v16h v; unsigned int u[8]; } fa0, fa1, fb0, fb1;
#pragma unroll
        for (int j = 0; j < 8; ++j) {
            // A 16x32 f16: lanes0-15 K=0..7 (v0..3), K=16..23 (v4..7); lanes16-31 +8
            int ka = (j < 4) ? (8 * hl + 2 * j) : (16 + 8 * hl + 2 * (j - 4));
            // B 32x16 f16: same K order as A (lanes0-15 K=0..7,16..23; lanes16-31 +8), packed pairs
            int kb = (j < 4) ? (8 * hl + 2 * j) : (16 + 8 * hl + 2 * (j - 4));
            fa0.u[j] = *(const unsigned int*)&As[buf][am0][ka];
            fa1.u[j] = *(const unsigned int*)&As[buf][am1][ka];
            fb0.u[j] = *(const unsigned int*)&Bs[buf][bn0][kb];
            fb1.u[j] = *(const unsigned int*)&Bs[buf][bn1][kb];
        }
        acc00 = __builtin_amdgcn_wmma_f32_16x16x32_f16(false, fa0.v, false, fb0.v, (short)0, acc00, false, false);
        acc01 = __builtin_amdgcn_wmma_f32_16x16x32_f16(false, fa0.v, false, fb1.v, (short)0, acc01, false, false);
        acc10 = __builtin_amdgcn_wmma_f32_16x16x32_f16(false, fa1.v, false, fb0.v, (short)0, acc10, false, false);
        acc11 = __builtin_amdgcn_wmma_f32_16x16x32_f16(false, fa1.v, false, fb1.v, (short)0, acc11, false, false);

        if (kk + 1 < nk) store_tile((kk + 1) & 1, rnxt);          // fill other buffer
        __syncthreads();
    }

    // ---- epilogue: D layout M = v + 8*hl, N = lane&15 ----
#pragma unroll
    for (int ms = 0; ms < 2; ++ms) {
#pragma unroll
        for (int ns = 0; ns < 2; ++ns) {
            const v8f* pacc = (ms == 0) ? (ns == 0 ? &acc00 : &acc01)
                                        : (ns == 0 ? &acc10 : &acc11);
            int gn = n0 + wn * 32 + ns * 16 + lr;
#pragma unroll
            for (int v = 0; v < 8; ++v) {
                int gm = m0 + wm * 32 + ms * 16 + 8 * hl + v;
                if (gm < M && gn < N) {
                    float val = (*pacc)[v];
                    if (bias) val += bias[gn];
                    if (op == OP_RELU)          val = fmaxf(val, 0.f);
                    else if (op == OP_SILU)     val = siluf_(val);
                    else if (op == OP_SOFTPLUS) val = softplusf_(val);
                    if (res) val += res[(long)gm * ldc + gn];
                    C[(long)gm * ldc + gn] = val;
                }
            }
        }
    }
}

// ---------------------------------------------------------------------------
// Embedding gather + positional add: out[b,s,:] = emb[tok[b,s],:] + pos[s,:]
// ---------------------------------------------------------------------------
__global__ __launch_bounds__(256)
void k_embed(const int* __restrict__ tok, const float* __restrict__ emb,
             const float* __restrict__ pos, float* __restrict__ out)
{
    int row = blockIdx.x;          // b*64 + s
    int s = row & 63;
    int e = threadIdx.x;           // 0..255
    int id = tok[row];
    out[(long)row * 256 + e] = emb[(long)id * 256 + e] + pos[s * 256 + e];
}

// ---------------------------------------------------------------------------
// MHA for one (batch, head): S=64, hd=64.  256 threads = 8 waves; each wave
// processes 8 query rows; lanes split key/value columns (j = lane, lane+32).
// ---------------------------------------------------------------------------
__global__ __launch_bounds__(256)
void k_attn(const float* __restrict__ qkv, float* __restrict__ o)
{
    __shared__ float Qs[64][65], Ks[64][65], Vs[64][65];
    __shared__ float Ps[8][64];

    int b = blockIdx.x >> 2, h = blockIdx.x & 3;
    const float* base = qkv + (long)b * 64 * 768 + h * 64;
    int t = threadIdx.x;
    for (int it = 0; it < 16; ++it) {
        int idx = t + it * 256;            // 0..4095
        int r = idx >> 6, c = idx & 63;
        Qs[r][c] = base[r * 768 + c];
        Ks[r][c] = base[r * 768 + 256 + c];
        Vs[r][c] = base[r * 768 + 512 + c];
    }
    __syncthreads();

    int wv = t >> 5, lane = t & 31;
    for (int qq = 0; qq < 8; ++qq) {
        int qr = wv * 8 + qq;
        float s0 = 0.f, s1 = 0.f;
        for (int e = 0; e < 64; ++e) {
            float qe = Qs[qr][e];
            s0 += qe * Ks[lane][e];
            s1 += qe * Ks[lane + 32][e];
        }
        s0 *= 0.125f; s1 *= 0.125f;        // 1/sqrt(64)
        float mx = fmaxf(s0, s1);
        for (int off = 16; off > 0; off >>= 1) mx = fmaxf(mx, __shfl_xor(mx, off, 32));
        float e0 = __expf(s0 - mx), e1 = __expf(s1 - mx);
        float sum = e0 + e1;
        for (int off = 16; off > 0; off >>= 1) sum += __shfl_xor(sum, off, 32);
        float inv = 1.f / sum;
        Ps[wv][lane] = e0 * inv; Ps[wv][lane + 32] = e1 * inv;   // wave-private row
        float a0 = 0.f, a1 = 0.f;
        for (int j = 0; j < 64; ++j) {
            float p = Ps[wv][j];
            a0 += p * Vs[j][lane];
            a1 += p * Vs[j][lane + 32];
        }
        long orow = ((long)(b * 64 + qr)) * 256 + h * 64;
        o[orow + lane] = a0;
        o[orow + lane + 32] = a1;
    }
}

// ---------------------------------------------------------------------------
// xout = LayerNorm(xin (+ delta)) * g + b.  One block per row, D = 256 or 512.
// ---------------------------------------------------------------------------
__global__ __launch_bounds__(256)
void k_add_ln(const float* __restrict__ xin, const float* __restrict__ delta,
              const float* __restrict__ g, const float* __restrict__ bb,
              float* __restrict__ xout, int D)
{
    __shared__ float s1[256], s2[256];
    long row = blockIdx.x;
    int t = threadIdx.x;
    int per = D >> 8;                       // 1 or 2
    float vals[2] = {0.f, 0.f};
    float a = 0.f, sq = 0.f;
    for (int i = 0; i < per; ++i) {
        int e = t + (i << 8);
        float v = xin[row * D + e];
        if (delta) v += delta[row * D + e];
        vals[i] = v; a += v; sq += v * v;
    }
    s1[t] = a; s2[t] = sq;
    __syncthreads();
    for (int st = 128; st > 0; st >>= 1) {
        if (t < st) { s1[t] += s1[t + st]; s2[t] += s2[t + st]; }
        __syncthreads();
    }
    float mean = s1[0] / (float)D;
    float var  = s2[0] / (float)D - mean * mean;
    float inv  = rsqrtf(var + 1e-5f);
    for (int i = 0; i < per; ++i) {
        int e = t + (i << 8);
        xout[row * D + e] = (vals[i] - mean) * inv * g[e] + bb[e];
    }
}

// mean over sequence (S=64): pooled[b,e]
__global__ __launch_bounds__(256)
void k_meanpool(const float* __restrict__ xt, float* __restrict__ pooled)
{
    int b = blockIdx.x; int e = threadIdx.x;
    float s = 0.f;
    for (int t = 0; t < 64; ++t) s += xt[((long)(b * 64 + t)) * 256 + e];
    pooled[b * 256 + e] = s * (1.f / 64.f);
}

// x[b,l,:] = cond[b,:] + pos_motion[l,:]
__global__ __launch_bounds__(256)
void k_build_x(const float* __restrict__ cond, const float* __restrict__ posm,
               float* __restrict__ x)
{
    int row = blockIdx.x; int b = row >> 8; int l = row & 255;
    for (int i = 0; i < 2; ++i) {
        int e = threadIdx.x + (i << 8);
        x[(long)row * 512 + e] = cond[b * 512 + e] + posm[l * 512 + e];
    }
}

// causal depthwise conv1d (K=4) over x1 half of xz, + bias, + silu
__global__ __launch_bounds__(256)
void k_dwconv(const float* __restrict__ xz, const float* __restrict__ cw,
              const float* __restrict__ cb, float* __restrict__ x1c)
{
    int row = blockIdx.x; int b = row >> 8; int l = row & 255;
    for (int it = 0; it < 4; ++it) {
        int d = threadIdx.x + (it << 8);
        float acc = cb[d];
#pragma unroll
        for (int tt = 0; tt < 4; ++tt) {
            int ll = l - 3 + tt;
            if (ll >= 0) acc += cw[d * 4 + tt] * xz[((long)(b * 256 + ll)) * 2048 + d];
        }
        x1c[(long)row * 1024 + d] = siluf_(acc);
    }
}

// selective scan: one thread per (b, d); 16 fp32 states in registers.
__global__ __launch_bounds__(256)
void k_scan(const float* __restrict__ x1c, const float* __restrict__ dtb,
            const float* __restrict__ dbl, const float* __restrict__ xz,
            const float* __restrict__ Alog, const float* __restrict__ Dp,
            float* __restrict__ y)
{
    int gid = blockIdx.x * 256 + threadIdx.x;     // b*1024 + d
    int b = gid >> 10, d = gid & 1023;
    float Ar[16], h[16];
#pragma unroll
    for (int n = 0; n < 16; ++n) { Ar[n] = -__expf(Alog[d * 16 + n]); h[n] = 0.f; }
    float Dv = Dp[d];
    long rb = (long)b * 256;
    for (int l = 0; l < 256; ++l) {
        long idx = rb + l;
        float dt = dtb[idx * 1024 + d];
        float xv = x1c[idx * 1024 + d];
        const float* bc = dbl + idx * 64;         // [32:48)=B, [48:64)=C (uniform in block)
        float dx = dt * xv;
        float yv = 0.f;
#pragma unroll
        for (int n = 0; n < 16; ++n) {
            float dA = __expf(dt * Ar[n]);
            h[n] = dA * h[n] + dx * bc[32 + n];
            yv  += h[n] * bc[48 + n];
        }
        yv += xv * Dv;
        float zv = xz[idx * 2048 + 1024 + d];
        yv *= zv * sigmoidf_(zv);                 // y * silu(z)
        y[idx * 1024 + d] = yv;
    }
}

// length_pred[b] = sigmoid(cond[b] . len_w + len_b) * 256
__global__ __launch_bounds__(256)
void k_lenpred(const float* __restrict__ cond, const float* __restrict__ lw,
               const float* __restrict__ lb, float* __restrict__ out)
{
    int t = threadIdx.x;
    if (t < 16) {
        float s = lb[0];
        for (int e = 0; e < 512; ++e) s += cond[t * 512 + e] * lw[e];
        out[t] = 256.f * sigmoidf_(s);
    }
}

// ---------------------------------------------------------------------------
extern "C" void kernel_launch(void* const* d_in, const int* in_sizes, int n_in,
                              void* d_out, int out_size, void* d_ws, size_t ws_size,
                              hipStream_t stream)
{
    (void)in_sizes; (void)n_in; (void)out_size; (void)ws_size;

    const int*   tok        = (const int*)  d_in[0];
    const float* emb_w      = (const float*)d_in[1];
    const float* pos_text   = (const float*)d_in[2];
    const float* qkv_w      = (const float*)d_in[3];
    const float* qkv_b      = (const float*)d_in[4];
    const float* attn_out_w = (const float*)d_in[5];
    const float* attn_out_b = (const float*)d_in[6];
    const float* lin1_w     = (const float*)d_in[7];
    const float* lin1_b     = (const float*)d_in[8];
    const float* lin2_w     = (const float*)d_in[9];
    const float* lin2_b     = (const float*)d_in[10];
    const float* ln1_g      = (const float*)d_in[11];
    const float* ln1_b      = (const float*)d_in[12];
    const float* ln2_g      = (const float*)d_in[13];
    const float* ln2_b      = (const float*)d_in[14];
    const float* cond_w     = (const float*)d_in[15];
    const float* cond_b     = (const float*)d_in[16];
    const float* pos_motion = (const float*)d_in[17];
    const float* in_w       = (const float*)d_in[18];
    const float* conv_w     = (const float*)d_in[19];
    const float* conv_b     = (const float*)d_in[20];
    const float* xp_w       = (const float*)d_in[21];
    const float* dt_w       = (const float*)d_in[22];
    const float* dt_b       = (const float*)d_in[23];
    const float* A_log      = (const float*)d_in[24];
    const float* Dp         = (const float*)d_in[25];
    const float* out_w      = (const float*)d_in[26];
    const float* norm_g     = (const float*)d_in[27];
    const float* norm_b     = (const float*)d_in[28];
    const float* len_w      = (const float*)d_in[29];
    const float* len_b      = (const float*)d_in[30];
    const float* o1_w       = (const float*)d_in[31];
    const float* o1_b       = (const float*)d_in[32];
    const float* o2_w       = (const float*)d_in[33];
    const float* o2_b       = (const float*)d_in[34];

    // ---- workspace layout (floats) ----
    float* ws     = (float*)d_ws;
    float* x      = ws;                       // [4096,512]
    float* xn     = x    + 2097152;           // [4096,512]
    float* xz     = xn   + 2097152;           // [4096,2048]
    float* x1c    = xz   + 8388608;           // [4096,1024]
    float* dblb   = x1c  + 4194304;           // [4096,64]
    float* dtbuf  = dblb + 262144;            // [4096,1024]
    float* yb     = dtbuf+ 4194304;           // [4096,1024]
    float* cond   = yb   + 4194304;           // [16,512]
    float* pooled = cond + 8192;              // [16,256]
    // text-phase buffers alias into xz region (text finishes before Mamba)
    float* xtext = xz;                        // [1024,256]
    float* qkvb  = xz + 262144;               // [1024,768]
    float* ao    = xz + 1048576;              // [1024,256]
    float* pd    = xz + 1310720;              // [1024,256]
    float* ffh   = xz + 1572864;              // [1024,2048]
    float* o1buf = xz;                        // [4096,512] (after Mamba)

    float* motion   = (float*)d_out;          // [4096,263]
    float* len_pred = motion + 16 * 256 * 263;

    auto gemm = [&](const float* Ap, const float* Wp, const float* bp, const float* rp,
                    float* Cp, int M, int N, int K, int lda, int ldw, int ldc, int op) {
        dim3 g((N + 63) / 64, (M + 127) / 128);
        k_gemm<<<g, 256, 0, stream>>>(Ap, Wp, bp, rp, Cp, M, N, K, lda, ldw, ldc, op);
    };

    // ================= text encoder =================
    k_embed<<<1024, 256, 0, stream>>>(tok, emb_w, pos_text, xtext);
    for (int l = 0; l < 2; ++l) {
        gemm(xtext, qkv_w + l * 196608, qkv_b + l * 768, nullptr,
             qkvb, 1024, 768, 256, 256, 256, 768, OP_NONE);
        k_attn<<<64, 256, 0, stream>>>(qkvb, ao);
        gemm(ao, attn_out_w + l * 65536, attn_out_b + l * 256, nullptr,
             pd, 1024, 256, 256, 256, 256, 256, OP_NONE);
        k_add_ln<<<1024, 256, 0, stream>>>(xtext, pd, ln1_g + l * 256, ln1_b + l * 256, xtext, 256);
        gemm(xtext, lin1_w + l * 524288, lin1_b + l * 2048, nullptr,
             ffh, 1024, 2048, 256, 256, 256, 2048, OP_RELU);
        gemm(ffh, lin2_w + l * 524288, lin2_b + l * 256, nullptr,
             pd, 1024, 256, 2048, 2048, 2048, 256, OP_NONE);
        k_add_ln<<<1024, 256, 0, stream>>>(xtext, pd, ln2_g + l * 256, ln2_b + l * 256, xtext, 256);
    }
    k_meanpool<<<16, 256, 0, stream>>>(xtext, pooled);
    gemm(pooled, cond_w, cond_b, nullptr, cond, 16, 512, 256, 256, 256, 512, OP_NONE);
    k_build_x<<<4096, 256, 0, stream>>>(cond, pos_motion, x);

    // ================= Mamba stack =================
    for (int i = 0; i < 6; ++i) {
        k_add_ln<<<4096, 256, 0, stream>>>(x, nullptr, norm_g + i * 512, norm_b + i * 512, xn, 512);
        gemm(xn, in_w + (long)i * 1048576, nullptr, nullptr,
             xz, 4096, 2048, 512, 512, 512, 2048, OP_NONE);
        k_dwconv<<<4096, 256, 0, stream>>>(xz, conv_w + i * 4096, conv_b + i * 1024, x1c);
        gemm(x1c, xp_w + i * 65536, nullptr, nullptr,
             dblb, 4096, 64, 1024, 1024, 1024, 64, OP_NONE);
        gemm(dblb, dt_w + i * 32768, dt_b + i * 1024, nullptr,
             dtbuf, 4096, 1024, 32, 64, 32, 1024, OP_SOFTPLUS);
        k_scan<<<64, 256, 0, stream>>>(x1c, dtbuf, dblb, xz,
                                       A_log + i * 16384, Dp + i * 1024, yb);
        gemm(yb, out_w + (long)i * 524288, nullptr, x,
             x, 4096, 512, 1024, 1024, 1024, 512, OP_NONE);   // residual fused
    }

    // ================= heads =================
    k_lenpred<<<1, 256, 0, stream>>>(cond, len_w, len_b, len_pred);
    gemm(x, o1_w, o1_b, nullptr, o1buf, 4096, 512, 512, 512, 512, 512, OP_SILU);
    gemm(o1buf, o2_w, o2_b, nullptr, motion, 4096, 263, 512, 512, 512, 263, OP_NONE);
}
